// NodeModel_19705309954368
// MI455X (gfx1250) — matrix operand order repaired and checked by reading, not compiled, not run on hardware
//
#include <hip/hip_runtime.h>

typedef __attribute__((ext_vector_type(16))) _Float16 v16h;
typedef __attribute__((ext_vector_type(8)))  _Float16 v8h;
typedef __attribute__((ext_vector_type(8)))  float    v8f;

#define LRELU_SLOPE 0.1f
#define WAVES_PER_BLOCK 4   // 128 threads, 4 wave32 -> 64 rows per block

// ---------------- small helpers ----------------

__global__ void k_zero(float* __restrict__ p, int n) {
  int i = blockIdx.x * blockDim.x + threadIdx.x;
  if (i < n) p[i] = 0.0f;
}

// column sums / sums-of-squares of edge-level concat [x[row] | edge_attr] (64 cols)
__global__ void k_edge_stats(const float* __restrict__ x, const int* __restrict__ row,
                             const float* __restrict__ ea, float* __restrict__ stats, int E) {
  int c = threadIdx.x;            // blockDim.x == 64
  float s = 0.f, q = 0.f;
  for (int r = blockIdx.x; r < E; r += gridDim.x) {
    float v = (c < 32) ? x[(size_t)row[r] * 32 + c] : ea[(size_t)r * 32 + (c - 32)];
    s += v; q += v * v;
  }
  atomicAdd(&stats[c], s);
  atomicAdd(&stats[64 + c], q);
}

// column sums / sums-of-squares of node-level concat [x | agg] (96 cols)
__global__ void k_node_stats(const float* __restrict__ x, const float* __restrict__ agg,
                             float* __restrict__ stats, int N) {
  int c = threadIdx.x;            // blockDim.x == 96
  float s = 0.f, q = 0.f;
  for (int r = blockIdx.x; r < N; r += gridDim.x) {
    float v = (c < 32) ? x[(size_t)r * 32 + c] : agg[(size_t)r * 64 + (c - 32)];
    s += v; q += v * v;
  }
  atomicAdd(&stats[c], s);
  atomicAdd(&stats[96 + c], q);
}

// (sum, sumsq) -> (scale, shift) so BN(v) = v*scale + shift
__global__ void k_finalize(float* __restrict__ stats, const float* __restrict__ g,
                           const float* __restrict__ b, float inv_count, int C) {
  int c = threadIdx.x;
  if (c >= C) return;
  float mean = stats[c] * inv_count;
  float var  = stats[C + c] * inv_count - mean * mean;
  float rstd = rsqrtf(var + 1e-5f);
  float sc   = g[c] * rstd;
  stats[c]     = sc;
  stats[C + c] = b[c] - mean * sc;
}

// ---------------- fused BN -> GEMM(WMMA f16) -> bias -> (LReLU) -> out + stats ----------------
// ASRC : 0 = edge concat (x[gatherIdx[r]] | srcB(edge_attr)), 1 = f16 buffer, 2 = node concat (x | agg)
// OMODE: 0 = store f16 + accumulate output column stats, 1 = scatter-add f32 into scatterDst,
//        2 = store f32 (final output)
template <int IW, int OW, int ASRC, bool LRELU, int OMODE>
__global__ void __launch_bounds__(128)
k_layer(const float* __restrict__ srcX,      // ASRC 0/2: first 32 cols (x)
        const float* __restrict__ srcB,      // ASRC 0: edge_attr (stride 32); ASRC 2: agg (stride 64)
        const _Float16* __restrict__ srcH,   // ASRC 1: f16 activations (stride IW)
        const int* __restrict__ gatherIdx,   // ASRC 0
        const float* __restrict__ ss,        // [2*IW] BN scale | shift
        const float* __restrict__ W,         // [IW*OW] f32 row-major (k-major)
        const float* __restrict__ bias,      // [OW]
        _Float16* __restrict__ outH,         // OMODE 0 (may alias srcH; preload-then-store keeps it safe)
        float* __restrict__ outStats,        // OMODE 0: [2*OW]
        float* __restrict__ scatterDst,      // OMODE 1
        const int* __restrict__ scatterIdx,  // OMODE 1
        float* __restrict__ outF32,          // OMODE 2
        int R) {
  // Weights staged TRANSPOSED in LDS: Wlds[n*IW + k] = W[k*OW + n]
  // -> each of a lane's two 8-element K runs of one output column is contiguous (16B aligned)
  __shared__ alignas(32) _Float16 Wlds[IW * OW];
  __shared__ float ssLds[2 * IW];

  const int tid = threadIdx.x;
  for (int i = tid; i < IW * OW; i += 128) {
    int n = i / IW, k = i - n * IW;
    Wlds[i] = (_Float16)W[k * OW + n];
  }
  for (int i = tid; i < 2 * IW; i += 128) ssLds[i] = ss[i];
  __syncthreads();

  const int wave = tid >> 5;
  const int lane = tid & 31;
  const int hl   = lane >> 4;    // lane half (0/1)
  const int l16  = lane & 15;
  const int rowbase = blockIdx.x * (16 * WAVES_PER_BLOCK) + wave * 16;
  if (rowbase >= R) return;      // wave-uniform

  const bool fullTile = (rowbase + 16 <= R);   // wave-uniform (always true when R%16==0)
  const int r  = rowbase + l16;
  const bool rv = r < R;
  int gidx = 0;
  if (ASRC == 0) gidx = rv ? gatherIdx[r] : 0;

  // 8-wide input chunk loader (chunk base cb is a multiple of 8; each chunk lies
  // entirely within one side of the concat, and all bases are 32B aligned)
  auto ldchunk = [&](int cb) -> v8f {
    v8f o;
    if (ASRC == 1) {
      v8h h = *(const v8h*)&srcH[(size_t)r * IW + cb];
#pragma unroll
      for (int i = 0; i < 8; ++i) o[i] = (float)h[i];
      return o;
    }
    const float* p;
    if (ASRC == 0)
      p = (cb < 32) ? &srcX[(size_t)gidx * 32 + cb] : &srcB[(size_t)r * 32 + (cb - 32)];
    else
      p = (cb < 32) ? &srcX[(size_t)r * 32 + cb] : &srcB[(size_t)r * 64 + (cb - 32)];
    o = *(const v8f*)p;
    return o;
  };

  // ----- preload + BN-normalize A fragments (ISA 16-bit A 16x32 layout) -----
  constexpr int KC = IW / 32;
  v16h A[KC];
  if (fullTile) {
#pragma unroll
    for (int kc = 0; kc < KC; ++kc) {
      const int c0b = kc * 32 + hl * 8;   // elements 0..7  -> K = hl*8 + i
      const int c1b = c0b + 16;           // elements 8..15 -> K = 16 + hl*8 + i
      v8f lo = ldchunk(c0b);
      v8f hi = ldchunk(c1b);
#pragma unroll
      for (int i = 0; i < 8; ++i) {
        float a0 = lo[i] * ssLds[c0b + i] + ssLds[IW + c0b + i];
        float a1 = hi[i] * ssLds[c1b + i] + ssLds[IW + c1b + i];
        A[kc][i]     = (_Float16)a0;
        A[kc][i + 8] = (_Float16)a1;
      }
    }
  } else {
#pragma unroll
    for (int kc = 0; kc < KC; ++kc) {
#pragma unroll
      for (int i = 0; i < 8; ++i) {
        const int c0 = kc * 32 + hl * 8 + i;
        const int c1 = c0 + 16;
        float v0 = 0.f, v1 = 0.f;
        if (rv) {
          if (ASRC == 0) {
            v0 = (c0 < 32) ? srcX[(size_t)gidx * 32 + c0] : srcB[(size_t)r * 32 + (c0 - 32)];
            v1 = (c1 < 32) ? srcX[(size_t)gidx * 32 + c1] : srcB[(size_t)r * 32 + (c1 - 32)];
          } else if (ASRC == 1) {
            v0 = (float)srcH[(size_t)r * IW + c0];
            v1 = (float)srcH[(size_t)r * IW + c1];
          } else {
            v0 = (c0 < 32) ? srcX[(size_t)r * 32 + c0] : srcB[(size_t)r * 64 + (c0 - 32)];
            v1 = (c1 < 32) ? srcX[(size_t)r * 32 + c1] : srcB[(size_t)r * 64 + (c1 - 32)];
          }
        }
        v0 = v0 * ssLds[c0] + ssLds[IW + c0];
        v1 = v1 * ssLds[c1] + ssLds[IW + c1];
        A[kc][i]     = (_Float16)v0;
        A[kc][i + 8] = (_Float16)v1;
      }
    }
  }

  // ----- GEMM over output 16-wide tiles -----
#pragma unroll
  for (int nt = 0; nt < OW / 16; ++nt) {
    v8f acc = {};
#pragma unroll
    for (int kc = 0; kc < KC; ++kc) {
      // 32x16 f16 B: lane = column l16, elements 0..7 -> K = kc*32 + hl*8 + i, 8..15 -> K = kc*32 + 16 + hl*8 + i
      // two 16B vector reads from transposed LDS
      const v8h Blo = *(const v8h*)&Wlds[(nt * 16 + l16) * IW + kc * 32 + hl * 8];
      const v8h Bhi = *(const v8h*)&Wlds[(nt * 16 + l16) * IW + kc * 32 + 16 + hl * 8];
      const v16h B = __builtin_shufflevector(Blo, Bhi, 0, 1, 2, 3, 4, 5, 6, 7, 8, 9, 10, 11, 12, 13, 14, 15);
      acc = __builtin_amdgcn_wmma_f32_16x16x32_f16(false, A[kc], false, B,
                                                   (short)0, acc, false, false);
    }

    // epilogue: C layout -> lane column = l16, rows M = hl*8 + i
    const int ncol = nt * 16 + l16;
    const float bv = bias[ncol];
    float s = 0.f, q = 0.f;
    if (fullTile) {
#pragma unroll
      for (int i = 0; i < 8; ++i) {
        const int gr = rowbase + hl * 8 + i;
        float v = acc[i] + bv;
        if (LRELU) v = (v > 0.f) ? v : LRELU_SLOPE * v;
        if (OMODE == 0) {
          outH[(size_t)gr * OW + ncol] = (_Float16)v;
          s += v; q += v * v;
        } else if (OMODE == 1) {
          atomicAdd(&scatterDst[(size_t)scatterIdx[gr] * OW + ncol], v);
        } else {
          outF32[(size_t)gr * OW + ncol] = v;
        }
      }
    } else {
#pragma unroll
      for (int i = 0; i < 8; ++i) {
        const int gr = rowbase + hl * 8 + i;
        if (gr < R) {
          float v = acc[i] + bv;
          if (LRELU) v = (v > 0.f) ? v : LRELU_SLOPE * v;
          if (OMODE == 0) {
            outH[(size_t)gr * OW + ncol] = (_Float16)v;
            s += v; q += v * v;
          } else if (OMODE == 1) {
            atomicAdd(&scatterDst[(size_t)scatterIdx[gr] * OW + ncol], v);
          } else {
            outF32[(size_t)gr * OW + ncol] = v;
          }
        }
      }
    }
    if (OMODE == 0) {
      atomicAdd(&outStats[ncol], s);
      atomicAdd(&outStats[OW + ncol], q);
    }
  }
}

// ---------------- launch ----------------

extern "C" void kernel_launch(void* const* d_in, const int* in_sizes, int n_in,
                              void* d_out, int out_size, void* d_ws, size_t ws_size,
                              hipStream_t stream) {
  (void)n_in; (void)out_size; (void)ws_size;

  const float* x   = (const float*)d_in[0];
  const int*   ei  = (const int*)d_in[1];
  const float* ea  = (const float*)d_in[2];
  // d_in[3] = u, d_in[4] = batch : unused by the node model
  const float *m1_g1 = (const float*)d_in[5],  *m1_b1 = (const float*)d_in[6];
  const float *m1_w1 = (const float*)d_in[7],  *m1_c1 = (const float*)d_in[8];
  const float *m1_g2 = (const float*)d_in[9],  *m1_b2 = (const float*)d_in[10];
  const float *m1_w2 = (const float*)d_in[11], *m1_c2 = (const float*)d_in[12];
  const float *m1_g3 = (const float*)d_in[13], *m1_b3 = (const float*)d_in[14];
  const float *m1_w3 = (const float*)d_in[15], *m1_c3 = (const float*)d_in[16];
  const float *m2_g1 = (const float*)d_in[17], *m2_b1 = (const float*)d_in[18];
  const float *m2_w1 = (const float*)d_in[19], *m2_c1 = (const float*)d_in[20];
  const float *m2_g2 = (const float*)d_in[21], *m2_b2 = (const float*)d_in[22];
  const float *m2_w2 = (const float*)d_in[23], *m2_c2 = (const float*)d_in[24];
  const float *m2_g3 = (const float*)d_in[25], *m2_b3 = (const float*)d_in[26];
  const float *m2_w3 = (const float*)d_in[27], *m2_c3 = (const float*)d_in[28];
  float* out = (float*)d_out;

  const int N = in_sizes[0] / 32;
  const int E = in_sizes[2] / 32;
  const int* rowI = ei;          // source nodes
  const int* colI = ei + E;      // destination nodes

  // ---- workspace layout ----
  char* ws = (char*)d_ws;
  float* s1 = (float*)ws;        // 128
  float* s2 = s1 + 128;          // 128
  float* s3 = s2 + 128;          // 128
  float* s4 = s3 + 128;          // 192
  float* s5 = s4 + 192;          // 192
  float* s6 = s5 + 192;          // 192  (total 960 floats)
  size_t off = 4096;
  float* agg = (float*)(ws + off);                         // N*64 f32 (atomic scatter target)
  off += ((size_t)N * 64 * sizeof(float) + 255) & ~(size_t)255;
  _Float16* hbuf = (_Float16*)(ws + off);                  // E*64 f16 (mlp1 activations, in-place L2)
  off += ((size_t)E * 64 * sizeof(_Float16) + 255) & ~(size_t)255;
  _Float16* gbuf = (_Float16*)(ws + off);                  // N*96 f16 (mlp2 activations, in-place L2)

  const float invE = 1.0f / (float)E;
  const float invN = 1.0f / (float)N;
  const int blkE = (E + 63) / 64;
  const int blkN = (N + 63) / 64;

  // zero stats + agg (agg is the scatter accumulator)
  k_zero<<<(960 + 255) / 256, 256, 0, stream>>>(s1, 960);
  k_zero<<<((N * 64) + 255) / 256, 256, 0, stream>>>(agg, N * 64);

  // ---- edge MLP (node_mlp_1) ----
  k_edge_stats<<<2048, 64, 0, stream>>>(x, rowI, ea, s1, E);
  k_finalize<<<1, 64, 0, stream>>>(s1, m1_g1, m1_b1, invE, 64);

  k_layer<64, 64, 0, true, 0><<<blkE, 128, 0, stream>>>(
      x, ea, nullptr, rowI, s1, m1_w1, m1_c1, hbuf, s2, nullptr, nullptr, nullptr, E);
  k_finalize<<<1, 64, 0, stream>>>(s2, m1_g2, m1_b2, invE, 64);

  k_layer<64, 64, 1, true, 0><<<blkE, 128, 0, stream>>>(
      nullptr, nullptr, hbuf, nullptr, s2, m1_w2, m1_c2, hbuf, s3, nullptr, nullptr, nullptr, E);
  k_finalize<<<1, 64, 0, stream>>>(s3, m1_g3, m1_b3, invE, 64);

  k_layer<64, 64, 1, false, 1><<<blkE, 128, 0, stream>>>(
      nullptr, nullptr, hbuf, nullptr, s3, m1_w3, m1_c3, nullptr, nullptr, agg, colI, nullptr, E);

  // ---- node MLP (node_mlp_2) ----
  k_node_stats<<<2048, 96, 0, stream>>>(x, agg, s4, N);
  k_finalize<<<1, 96, 0, stream>>>(s4, m2_g1, m2_b1, invN, 96);

  k_layer<96, 96, 2, true, 0><<<blkN, 128, 0, stream>>>(
      x, agg, nullptr, nullptr, s4, m2_w1, m2_c1, gbuf, s5, nullptr, nullptr, nullptr, N);
  k_finalize<<<1, 96, 0, stream>>>(s5, m2_g2, m2_b2, invN, 96);

  k_layer<96, 96, 1, true, 0><<<blkN, 128, 0, stream>>>(
      nullptr, nullptr, gbuf, nullptr, s5, m2_w2, m2_c2, gbuf, s6, nullptr, nullptr, nullptr, N);
  k_finalize<<<1, 96, 0, stream>>>(s6, m2_g3, m2_b3, invN, 96);

  k_layer<96, 32, 1, false, 2><<<blkN, 128, 0, stream>>>(
      nullptr, nullptr, gbuf, nullptr, s6, m2_w3, m2_c3, nullptr, nullptr, nullptr, nullptr, out, N);
}
